// GatedSSMBlock_790273982961
// MI455X (gfx1250) — compile-verified, matrix operand order checked by reading, not run on hardware
//
#include <hip/hip_runtime.h>
#include <hip/hip_bf16.h>

// ---------- CDNA5 WMMA types ----------
typedef __attribute__((ext_vector_type(16))) __bf16        v16bf;
typedef __attribute__((ext_vector_type(8)))  float         v8f;
typedef __attribute__((ext_vector_type(4)))  unsigned int  u32x4;
typedef __attribute__((ext_vector_type(8)))  unsigned int  u32x8;

struct __align__(8) bf16x4 { __bf16 x, y, z, w; };

// Load one 16-bit WMMA A/B fragment (16 bf16 per lane) from a row-major
// (rows x K) matrix.  Per ISA 7.12.2: lane L holds row (L&15); lanes 0-15
// carry K = [0..7] and [16..23], lanes 16-31 carry K = [8..15] and [24..31].
// Caller passes p already offset by (row)*K + (lane>>4)*8 + k0, so the two
// 16-byte chunks are at p and p+16 elements.
static __device__ inline v16bf load_frag(const __bf16* p) {
    u32x4 lo = *(const u32x4*)p;
    u32x4 hi = *(const u32x4*)(p + 16);
    u32x8 w  = __builtin_shufflevector(lo, hi, 0, 1, 2, 3, 4, 5, 6, 7);
    return __builtin_bit_cast(v16bf, w);
}

// ---------- 1) weight transpose + fp32->bf16 cast:  W[K][N] -> WT[N][K] ----------
__global__ void transpose_cast_kernel(const float* __restrict__ W,
                                      __bf16* __restrict__ WT,
                                      int K, int N) {
    int idx = blockIdx.x * blockDim.x + threadIdx.x;
    if (idx >= K * N) return;
    int k = idx / N;
    int n = idx - k * N;                 // consecutive threads -> coalesced read
    WT[(size_t)n * K + k] = (__bf16)W[idx];
}

// ---------- 2) fused LayerNorm -> bf16 z ----------
__global__ void ln_kernel(const float* __restrict__ x,
                          const float* __restrict__ gamma,
                          const float* __restrict__ beta,
                          __bf16* __restrict__ z, int D) {
    const int row = blockIdx.x;
    const int tid = threadIdx.x;               // 256 threads, 4 floats each (D=1024)
    const float4 v = ((const float4*)(x + (size_t)row * D))[tid];

    float s  = v.x + v.y + v.z + v.w;
    float s2 = v.x * v.x + v.y * v.y + v.z * v.z + v.w * v.w;
    #pragma unroll
    for (int off = 16; off > 0; off >>= 1) {   // wave32 reduction
        s  += __shfl_xor(s,  off, 32);
        s2 += __shfl_xor(s2, off, 32);
    }
    __shared__ float sh[16];
    const int w = tid >> 5, lane = tid & 31;
    if (lane == 0) { sh[w] = s; sh[8 + w] = s2; }
    __syncthreads();
    s = 0.f; s2 = 0.f;
    #pragma unroll
    for (int i = 0; i < 8; ++i) { s += sh[i]; s2 += sh[8 + i]; }

    const float invD = 1.f / (float)D;
    const float mu   = s * invD;
    const float var  = s2 * invD - mu * mu;
    const float rstd = rsqrtf(var + 1e-5f);

    const float4 g4 = ((const float4*)gamma)[tid];
    const float4 b4 = ((const float4*)beta)[tid];
    bf16x4 o;
    o.x = (__bf16)((v.x - mu) * rstd * g4.x + b4.x);
    o.y = (__bf16)((v.y - mu) * rstd * g4.y + b4.y);
    o.z = (__bf16)((v.z - mu) * rstd * g4.z + b4.z);
    o.w = (__bf16)((v.w - mu) * rstd * g4.w + b4.w);
    ((bf16x4*)(z + (size_t)row * D))[tid] = o;
}

// ---------- 3/5) bf16 WMMA GEMM:  C[M][N] = A[M][K] * BT[N][K]^T ----------
// Block = 8 waves (4 M x 2 N); wave tile 32x64 = 2x4 WMMA frags; block tile
// 128x128.  Explicit unroll-by-2 k-loop with two statically named fragment
// sets (ping/pong) so the allocator can keep all 8 accumulator octets and
// both load sets in fixed even-aligned homes (avoids v_mov rotation + the
// WMMA->VALU hazard v_nops it forces).  8 WMMAs per 12 b128 loads per
// k-step, ~21 FLOP/byte of cache-side traffic; ~190 VGPRs, under the
// 256-VGPR spill cliff.
// EPI==0: ug epilogue (bias + split u / sigmoid(g), bf16 out)
// EPI==1: out epilogue (bias + residual, fp32 out)
template <int EPI>
__global__ void wmma_gemm_kernel(const __bf16* __restrict__ A,
                                 const __bf16* __restrict__ BT,
                                 int Mdim, int Ndim, int Kdim,
                                 const float* __restrict__ bias,
                                 const float* __restrict__ resid,
                                 float* __restrict__ outF,
                                 __bf16* __restrict__ u_buf,
                                 __bf16* __restrict__ g_buf, int Dh) {
    const int lane = threadIdx.x & 31;
    const int wv   = threadIdx.x >> 5;
    const int m    = lane & 15;     // row (A) / col (B) within 16
    const int kh   = lane >> 4;     // K-half selector
    const int rowBase = blockIdx.x * 128 + (wv >> 1) * 32;
    const int colBase = blockIdx.y * 128 + (wv & 1)  * 64;

    const __bf16* ap[2];
    const __bf16* bp[4];
    #pragma unroll
    for (int i = 0; i < 2; ++i)
        ap[i] = A  + (size_t)(rowBase + 16 * i + m) * Kdim + kh * 8;
    #pragma unroll
    for (int j = 0; j < 4; ++j)
        bp[j] = BT + (size_t)(colBase + 16 * j + m) * Kdim + kh * 8;

    v8f acc[2][4] = {};
    #pragma unroll 1
    for (int k0 = 0; k0 < Kdim; k0 += 64) {
        // ping set @ k0
        v16bf a0 = load_frag(ap[0] + k0);
        v16bf a1 = load_frag(ap[1] + k0);
        v16bf b0 = load_frag(bp[0] + k0);
        v16bf b1 = load_frag(bp[1] + k0);
        v16bf b2 = load_frag(bp[2] + k0);
        v16bf b3 = load_frag(bp[3] + k0);
        // pong set @ k0+32
        v16bf c0 = load_frag(ap[0] + k0 + 32);
        v16bf c1 = load_frag(ap[1] + k0 + 32);
        v16bf d0 = load_frag(bp[0] + k0 + 32);
        v16bf d1 = load_frag(bp[1] + k0 + 32);
        v16bf d2 = load_frag(bp[2] + k0 + 32);
        v16bf d3 = load_frag(bp[3] + k0 + 32);

        acc[0][0] = __builtin_amdgcn_wmma_f32_16x16x32_bf16(false, a0, false, b0, (short)0, acc[0][0], false, false);
        acc[0][1] = __builtin_amdgcn_wmma_f32_16x16x32_bf16(false, a0, false, b1, (short)0, acc[0][1], false, false);
        acc[0][2] = __builtin_amdgcn_wmma_f32_16x16x32_bf16(false, a0, false, b2, (short)0, acc[0][2], false, false);
        acc[0][3] = __builtin_amdgcn_wmma_f32_16x16x32_bf16(false, a0, false, b3, (short)0, acc[0][3], false, false);
        acc[1][0] = __builtin_amdgcn_wmma_f32_16x16x32_bf16(false, a1, false, b0, (short)0, acc[1][0], false, false);
        acc[1][1] = __builtin_amdgcn_wmma_f32_16x16x32_bf16(false, a1, false, b1, (short)0, acc[1][1], false, false);
        acc[1][2] = __builtin_amdgcn_wmma_f32_16x16x32_bf16(false, a1, false, b2, (short)0, acc[1][2], false, false);
        acc[1][3] = __builtin_amdgcn_wmma_f32_16x16x32_bf16(false, a1, false, b3, (short)0, acc[1][3], false, false);

        acc[0][0] = __builtin_amdgcn_wmma_f32_16x16x32_bf16(false, c0, false, d0, (short)0, acc[0][0], false, false);
        acc[0][1] = __builtin_amdgcn_wmma_f32_16x16x32_bf16(false, c0, false, d1, (short)0, acc[0][1], false, false);
        acc[0][2] = __builtin_amdgcn_wmma_f32_16x16x32_bf16(false, c0, false, d2, (short)0, acc[0][2], false, false);
        acc[0][3] = __builtin_amdgcn_wmma_f32_16x16x32_bf16(false, c0, false, d3, (short)0, acc[0][3], false, false);
        acc[1][0] = __builtin_amdgcn_wmma_f32_16x16x32_bf16(false, c1, false, d0, (short)0, acc[1][0], false, false);
        acc[1][1] = __builtin_amdgcn_wmma_f32_16x16x32_bf16(false, c1, false, d1, (short)0, acc[1][1], false, false);
        acc[1][2] = __builtin_amdgcn_wmma_f32_16x16x32_bf16(false, c1, false, d2, (short)0, acc[1][2], false, false);
        acc[1][3] = __builtin_amdgcn_wmma_f32_16x16x32_bf16(false, c1, false, d3, (short)0, acc[1][3], false, false);
    }

    // C fragment mapping (ISA 7.12.2): element r of lane l -> M = r + 8*(l>>4), N = l&15
    #pragma unroll
    for (int i = 0; i < 2; ++i) {
        #pragma unroll
        for (int j = 0; j < 4; ++j) {
            const int   col  = colBase + 16 * j + m;
            const float bcol = bias[col];
            #pragma unroll
            for (int r = 0; r < 8; ++r) {
                const int   row = rowBase + 16 * i + 8 * kh + r;
                const float val = acc[i][j][r] + bcol;
                if (EPI == 0) {
                    if (col < Dh)
                        u_buf[(size_t)row * Dh + col] = (__bf16)val;
                    else
                        g_buf[(size_t)row * Dh + (col - Dh)] =
                            (__bf16)(1.f / (1.f + __expf(-val)));
                } else {
                    const size_t o = (size_t)row * Ndim + col;
                    outF[o] = resid[o] + val;
                }
            }
        }
    }
}

// ---------- 4) per-channel gated scan:  h = g*h + (1-g)*tanh(A*h + u) ----------
__global__ void scan_kernel(const __bf16* __restrict__ u,
                            const __bf16* __restrict__ g,
                            const float* __restrict__ Avec,
                            __bf16* __restrict__ hs,
                            int L, int D) {
    const int idx = blockIdx.x * blockDim.x + threadIdx.x; // b*D + d
    const int b = idx / D;
    const int d = idx - b * D;
    const float a = Avec[d];
    float h = 0.f;
    size_t off = (size_t)b * L * D + d;                    // coalesced across d
    for (int t = 0; t < L; ++t, off += D) {
        const float uu = (float)u[off];
        const float gg = (float)g[off];
        h = gg * h + (1.f - gg) * tanhf(fmaf(a, h, uu));
        hs[off] = (__bf16)h;
    }
}

extern "C" void kernel_launch(void* const* d_in, const int* in_sizes, int n_in,
                              void* d_out, int out_size, void* d_ws, size_t ws_size,
                              hipStream_t stream) {
    const float* x     = (const float*)d_in[0];
    const float* gamma = (const float*)d_in[1];
    const float* beta  = (const float*)d_in[2];
    const float* Avec  = (const float*)d_in[3];
    const float* W_in  = (const float*)d_in[4];
    const float* b_in  = (const float*)d_in[5];
    const float* W_out = (const float*)d_in[6];
    const float* b_out = (const float*)d_in[7];
    float* out = (float*)d_out;

    const int Bb = 16, L = 2048, D = 1024;
    const int M  = Bb * L;       // 32768 rows
    const int N1 = 2 * D;        // 2048

    // workspace layout (~198 MiB):
    //  z / hs (aliased, bf16) | u (bf16) | g (bf16) | W_in^T (bf16) | W_out^T (bf16)
    __bf16* z_hs   = (__bf16*)d_ws;
    __bf16* u_buf  = z_hs  + (size_t)M * D;
    __bf16* g_buf  = u_buf + (size_t)M * D;
    __bf16* WT_in  = g_buf + (size_t)M * D;
    __bf16* WT_out = WT_in + (size_t)N1 * D;

    transpose_cast_kernel<<<(D * N1 + 255) / 256, 256, 0, stream>>>(W_in, WT_in, D, N1);
    transpose_cast_kernel<<<(D * D  + 255) / 256, 256, 0, stream>>>(W_out, WT_out, D, D);

    ln_kernel<<<M, 256, 0, stream>>>(x, gamma, beta, z_hs, D);

    wmma_gemm_kernel<0><<<dim3(M / 128, N1 / 128), 256, 0, stream>>>(
        z_hs, WT_in, M, N1, D, b_in, nullptr, nullptr, u_buf, g_buf, D);

    scan_kernel<<<(Bb * D) / 256, 256, 0, stream>>>(u_buf, g_buf, Avec, z_hs, L, D);

    wmma_gemm_kernel<1><<<dim3(M / 128, D / 128), 256, 0, stream>>>(
        z_hs, WT_out, M, D, D, b_out, x, out, nullptr, nullptr, D);
}
